// CompositionalFFN_6167573037667
// MI455X (gfx1250) — matrix operand order repaired and checked by reading, not compiled, not run on hardware
//
#include <hip/hip_runtime.h>
#include <math.h>

// ---------------- Types for WMMA fragments ----------------
typedef __attribute__((ext_vector_type(16))) __bf16 v16bf;
typedef __attribute__((ext_vector_type(8)))  __bf16 v8bf;
typedef __attribute__((ext_vector_type(8)))  float  v8f;

// Problem constants (match reference setup_inputs)
#define CB   2
#define CS   1024
#define CD   1024
#define CF   4096
#define CP   256
#define CR   64
#define CK   32           // TOP_K
#define KR   (CK * CR)    // 2048: contracted dim of the compose GEMMs
#define CM0  (CB * CS)    // 2048 token rows

// ---------------- Top-k + softmax (both routers in one launch) ----------------
__global__ void cffn_topk_softmax(const float* __restrict__ lg1,
                                  const float* __restrict__ lg2,
                                  int* __restrict__ idx1, float* __restrict__ w1,
                                  int* __restrict__ idx2, float* __restrict__ w2) {
  __shared__ float vals[CP];
  __shared__ int   sidx[CK];
  __shared__ float sval[CK];
  const float* lg = (blockIdx.x == 0) ? lg1 : lg2;
  int*   oi = (blockIdx.x == 0) ? idx1 : idx2;
  float* ow = (blockIdx.x == 0) ? w1 : w2;
  int t = threadIdx.x;
  if (t < CP) vals[t] = lg[t];
  __syncthreads();
  if (t == 0) {
    for (int k = 0; k < CK; ++k) {            // selection top-k, first-index ties
      int best = 0; float bv = vals[0];
      for (int p = 1; p < CP; ++p) { if (vals[p] > bv) { bv = vals[p]; best = p; } }
      sidx[k] = best; sval[k] = bv; vals[best] = -3.4e38f;
    }
    float m = sval[0], s = 0.f;               // softmax (TEMPERATURE = 1)
    for (int k = 0; k < CK; ++k) { sval[k] = __expf(sval[k] - m); s += sval[k]; }
    float inv = 1.f / s;
    for (int k = 0; k < CK; ++k) { oi[k] = sidx[k]; ow[k] = sval[k] * inv; }
  }
}

// ---------------- Gathers: build bf16 operands in WMMA-native layouts ----------------
// out[row, k*R + r] = w[k] * A[idx[k], row, r]   (A is (P, rows, R), contiguous in r)
__global__ void cffn_gather_scale_A(const float* __restrict__ A,
                                    const int* __restrict__ idx,
                                    const float* __restrict__ w,
                                    __bf16* __restrict__ out, int rows) {
  int row = blockIdx.x, k = blockIdx.y, r = threadIdx.x;
  int p = idx[k];
  float v = A[((size_t)p * rows + row) * CR + r] * w[k];
  out[(size_t)row * KR + k * CR + r] = (__bf16)v;
}

// out[n, k*R + r] = B[idx[k], r, n]   (B is (P, R, N)) -> stores B_cat^T row-major
__global__ void cffn_gather_BT(const float* __restrict__ Bsrc,
                               const int* __restrict__ idx,
                               __bf16* __restrict__ out, int N) {
  int n = blockIdx.x * 64 + threadIdx.x;     // coalesced reads over n
  int k = blockIdx.y;
  size_t pbase = (size_t)idx[k] * CR * N;
  for (int r = threadIdx.y; r < CR; r += 4) {
    float v = Bsrc[pbase + (size_t)r * N + n];
    out[(size_t)n * KR + k * CR + r] = (__bf16)v;
  }
}

__global__ void cffn_f32_to_bf16(const float* __restrict__ in,
                                 __bf16* __restrict__ out, int n) {
  int i = blockIdx.x * blockDim.x + threadIdx.x;
  if (i < n) out[i] = (__bf16)in[i];
}

// ---------------- bf16 WMMA GEMM: C(MxN) = A(MxK) * B(KxN), B given as B^T (NxK) ----------
__device__ __forceinline__ float cffn_gelu(float x) {
  float u = 0.7978845608028654f * (x + 0.044715f * x * x * x);
  return 0.5f * x * (1.0f + tanhf(u));
}

template <bool GELU, bool OUTF32>
__global__ __launch_bounds__(256)
void cffn_gemm_bf16_wmma(const __bf16* __restrict__ A, const __bf16* __restrict__ BT,
                         void* __restrict__ Cout, int M, int N, int K) {
  const int bm   = blockIdx.y * 128;
  const int bn   = blockIdx.x * 128;
  const int wave = threadIdx.x >> 5;        // 8 wave32s per block
  const int lane = threadIdx.x & 31;
  const int wm   = wave & 3;                // 4 waves along M (32 rows each)
  const int wn   = wave >> 2;               // 2 waves along N (64 cols each)
  const int l    = lane & 15;
  const int half = lane >> 4;

  const __bf16* Abase = A  + (size_t)(bm + wm * 32 + l) * K;   // + i*16*K rows
  const __bf16* Bbase = BT + (size_t)(bn + wn * 64 + l) * K;   // + j*16*K rows

  v8f acc[2][4] = {};

  for (int k0 = 0; k0 < K; k0 += 32) {
    v16bf a[2], b[4];
    // A 16x32 bf16 fragment: lanes 0-15 rows M..M+15 K{0..7,16..23}; lanes 16-31 K{8..15,24..31}
    #pragma unroll
    for (int i = 0; i < 2; ++i) {
      const __bf16* p = Abase + (size_t)i * 16 * K + k0 + half * 8;
      v8bf lo = *(const v8bf*)(p);
      v8bf hi = *(const v8bf*)(p + 16);
      #pragma unroll
      for (int t = 0; t < 8; ++t) { a[i][t] = lo[t]; a[i][t + 8] = hi[t]; }
    }
    // B 32x16 bf16 fragment from B^T rows: lanes 0-15 K{0..7,16..23}; lanes 16-31 K{8..15,24..31}
    #pragma unroll
    for (int j = 0; j < 4; ++j) {
      const __bf16* p = Bbase + (size_t)j * 16 * K + k0 + half * 8;
      v8bf lo = *(const v8bf*)(p);
      v8bf hi = *(const v8bf*)(p + 16);
      #pragma unroll
      for (int t = 0; t < 8; ++t) { b[j][t] = lo[t]; b[j][t + 8] = hi[t]; }
    }
    if (k0 + 32 < K) {                       // pull next K-slice toward L2/L0
      __builtin_prefetch(Abase + (k0 + 32), 0, 1);
      __builtin_prefetch(Bbase + (k0 + 32), 0, 1);
    }
    #pragma unroll
    for (int i = 0; i < 2; ++i)
      #pragma unroll
      for (int j = 0; j < 4; ++j)
        acc[i][j] = __builtin_amdgcn_wmma_f32_16x16x32_bf16(
            false, a[i], false, b[j], (short)0, acc[i][j], false, false);
  }

  // C 16x16 f32 frag: VGPR v -> row (half*8 + v), lane l -> col
  #pragma unroll
  for (int i = 0; i < 2; ++i) {
    #pragma unroll
    for (int j = 0; j < 4; ++j) {
      #pragma unroll
      for (int v = 0; v < 8; ++v) {
        int row = bm + wm * 32 + i * 16 + half * 8 + v;
        int col = bn + wn * 64 + j * 16 + l;
        float x = acc[i][j][v];
        if (GELU) x = cffn_gelu(x);
        if (OUTF32) ((float*)Cout)[(size_t)row * N + col] = x;
        else        ((__bf16*)Cout)[(size_t)row * N + col] = (__bf16)x;
      }
    }
  }
}

// ---------------- Host-side orchestration ----------------
extern "C" void kernel_launch(void* const* d_in, const int* in_sizes, int n_in,
                              void* d_out, int out_size, void* d_ws, size_t ws_size,
                              hipStream_t stream) {
  const float* x     = (const float*)d_in[0];
  const float* fc1_A = (const float*)d_in[1];
  const float* fc1_B = (const float*)d_in[2];
  const float* fc2_A = (const float*)d_in[3];
  const float* fc2_B = (const float*)d_in[4];
  const float* lg1   = (const float*)d_in[5];
  const float* lg2   = (const float*)d_in[6];

  char* ws = (char*)d_ws;
  const size_t MB = (size_t)1 << 20;
  int*   idx1 = (int*)(ws + 0);
  float* w1   = (float*)(ws + 128);
  int*   idx2 = (int*)(ws + 256);
  float* w2   = (float*)(ws + 384);
  // 20 MB scratch region, reused across phases (stream-ordered):
  __bf16* A1cat  = (__bf16*)(ws + 1024);             //  4 MB (1024 x 2048)
  __bf16* B1catT = (__bf16*)(ws + 1024 + 4  * MB);   // 16 MB (4096 x 2048)
  __bf16* A2cat  = (__bf16*)(ws + 1024);             // 16 MB (4096 x 2048), reuse
  __bf16* B2catT = (__bf16*)(ws + 1024 + 16 * MB);   //  4 MB (1024 x 2048)
  __bf16* H      = (__bf16*)(ws + 1024);             // 16 MB (2048 x 4096), reuse
  __bf16* W1T    = (__bf16*)(ws + 1024 + 20 * MB);   //  8 MB (4096 x 1024)
  __bf16* W2T    = (__bf16*)(ws + 1024 + 28 * MB);   //  8 MB (1024 x 4096)
  __bf16* XB     = (__bf16*)(ws + 1024 + 36 * MB);   //  4 MB (2048 x 1024)
  float*  OUT    = (float*)d_out;                    // 2048 x 1024 f32

  // 1) routing: top-32 + softmax for both layers
  cffn_topk_softmax<<<2, 256, 0, stream>>>(lg1, lg2, idx1, w1, idx2, w2);
  // 2) x -> bf16
  cffn_f32_to_bf16<<<(CM0 * CD) / 256, 256, 0, stream>>>(x, XB, CM0 * CD);
  // 3) layer-1 gathers (softmax weight folded into A side)
  cffn_gather_scale_A<<<dim3(CD, CK), CR, 0, stream>>>(fc1_A, idx1, w1, A1cat, CD);
  cffn_gather_BT<<<dim3(CF / 64, CK), dim3(64, 4), 0, stream>>>(fc1_B, idx1, B1catT, CF);
  // 4) W1^T (4096x1024) = B1cat^T (4096x2048) @ A1cat^T; BT operand = A1cat row-major
  cffn_gemm_bf16_wmma<false, false><<<dim3(CD / 128, CF / 128), 256, 0, stream>>>(
      B1catT, A1cat, W1T, CF, CD, KR);
  // 5) layer-2 gathers (reuse scratch)
  cffn_gather_scale_A<<<dim3(CF, CK), CR, 0, stream>>>(fc2_A, idx2, w2, A2cat, CF);
  cffn_gather_BT<<<dim3(CD / 64, CK), dim3(64, 4), 0, stream>>>(fc2_B, idx2, B2catT, CD);
  // 6) W2^T (1024x4096) = B2cat^T (1024x2048) @ A2cat^T; BT operand = A2cat row-major
  cffn_gemm_bf16_wmma<false, false><<<dim3(CF / 128, CD / 128), 256, 0, stream>>>(
      B2catT, A2cat, W2T, CD, CF, KR);
  // 7) H (2048x4096) = gelu(x @ W1); B operand W1 given as W1^T rows
  cffn_gemm_bf16_wmma<true, false><<<dim3(CF / 128, CM0 / 128), 256, 0, stream>>>(
      XB, W1T, H, CM0, CF, CD);
  // 8) OUT (2048x1024, f32) = H @ W2; B operand W2 given as W2^T rows
  cffn_gemm_bf16_wmma<false, true><<<dim3(CD / 128, CM0 / 128), 256, 0, stream>>>(
      H, W2T, OUT, CM0, CD, CF);
}
